// SinkhornSelfAttention_16157666968221
// MI455X (gfx1250) — compile-verified, matrix operand order checked by reading, not run on hardware
//
#include <hip/hip_runtime.h>

// ---------------------------------------------------------------------------
// Sinkhorn block-sparse self-attention for MI455X (gfx1250, wave32, WMMA).
// Heavy GEMMs + block attention use v_wmma_f32_16x16x32_bf16.
// GEMM stages tiles into LDS with gfx1250 async copies
// (global_load_async_to_lds_b128 / s_wait_asynccnt) and double-buffers.
// ---------------------------------------------------------------------------

typedef __bf16 bf16;
typedef __attribute__((ext_vector_type(16))) __bf16 v16bf;
typedef __attribute__((ext_vector_type(8)))  __bf16 v8bf;
typedef __attribute__((ext_vector_type(8)))  float  v8f;

#define S_LEN  4096
#define BATCH  4
#define DMODEL 1024
#define NBLK   16
#define BS     256
#define NHEAD  16
#define HDIM   64
#define NROWS  (S_LEN * BATCH)          // 16384 GEMM rows (r = s*B + b)
#define SCALEF 0.125f                   // HDIM^-0.5
#define RSCALE 0.03125f                 // DMODEL^-0.5

// ---- CDNA5 bf16 fragment load: 16x32 tile, row-major, per ISA layout -------
// lane 0-15 : row = lane,    K = [0..7] (v0-3) and [16..23] (v4-7)
// lane16-31 : row = lane-16, K = [8..15]       and [24..31]
__device__ __forceinline__ v16bf frag_ld(const bf16* rowbase, int ld, int lane) {
    const bf16* p = rowbase + (size_t)(lane & 15) * ld + ((lane >> 4) << 3);
    v8bf c0 = *(const v8bf*)(p);
    v8bf c1 = *(const v8bf*)(p + 16);
    return __builtin_shufflevector(c0, c1, 0,1,2,3,4,5,6,7,8,9,10,11,12,13,14,15);
}

// ---- gfx1250 async copy: 16 bytes per lane, global -> LDS (ASYNCcnt) -------
__device__ __forceinline__ void async_cp16(unsigned lds_off, const void* gptr) {
    unsigned long long ga = (unsigned long long)gptr;
    asm volatile("global_load_async_to_lds_b128 %0, %1, off"
                 :: "v"(lds_off), "v"(ga) : "memory");
}

// ---------------------------------------------------------------------------
// 1) Block summaries: xb[b][n][d] = mean_i x[(n*256+i), b, d]
// ---------------------------------------------------------------------------
__global__ void block_mean_kernel(const float* __restrict__ x, float* __restrict__ xb) {
    int b = blockIdx.x >> 4, n = blockIdx.x & 15;
#pragma unroll
    for (int dd = 0; dd < 4; ++dd) {
        int d = threadIdx.x + (dd << 8);
        float s = 0.f;
        for (int i = 0; i < BS; ++i)
            s += x[(size_t)(((n << 8) + i) * BATCH + b) * DMODEL + d];
        xb[(size_t)(b * NBLK + n) * DMODEL + d] = s * (1.0f / 256.0f);
    }
}

// ---------------------------------------------------------------------------
// 2) q_blk / k_blk projections of block summaries (tiny GEMV, f32 VALU)
// ---------------------------------------------------------------------------
__global__ void blk_proj_kernel(const float* __restrict__ xb,
                                const float* __restrict__ Wq, const float* __restrict__ bq,
                                const float* __restrict__ Wk, const float* __restrict__ bk,
                                float* __restrict__ qb, float* __restrict__ kb) {
    int bn = blockIdx.x;                       // (b*16+n)
    __shared__ float xrow[DMODEL];
    for (int i = threadIdx.x; i < DMODEL; i += 256) xrow[i] = xb[(size_t)bn * DMODEL + i];
    __syncthreads();
#pragma unroll
    for (int jj = 0; jj < 4; ++jj) {
        int j = threadIdx.x + (jj << 8);
        float sq = 0.f, sk = 0.f;
        const float* wq = Wq + (size_t)j * DMODEL;
        const float* wk = Wk + (size_t)j * DMODEL;
        for (int d = 0; d < DMODEL; ++d) { float xv = xrow[d]; sq += xv * wq[d]; sk += xv * wk[d]; }
        qb[(size_t)bn * DMODEL + j] = sq + bq[j];
        kb[(size_t)bn * DMODEL + j] = sk + bk[j];
    }
}

// ---------------------------------------------------------------------------
// 3) logits + 5 Sinkhorn iterations + row argmax -> perm[b][n]
// ---------------------------------------------------------------------------
__global__ void sinkhorn_route_kernel(const float* __restrict__ qb,
                                      const float* __restrict__ kb,
                                      int* __restrict__ perm) {
    int b = blockIdx.x;
    int n = threadIdx.x >> 4, m = threadIdx.x & 15;
    const float* qr = qb + (size_t)(b * NBLK + n) * DMODEL;
    const float* kr = kb + (size_t)(b * NBLK + m) * DMODEL;
    float s = 0.f;
    for (int d = 0; d < DMODEL; ++d) s += qr[d] * kr[d];
    float la = s * RSCALE;
    __shared__ float Mt[16][17];
    for (int it = 0; it < 5; ++it) {
        Mt[n][m] = la; __syncthreads();
        float mx = -1e30f;
        for (int j = 0; j < 16; ++j) mx = fmaxf(mx, Mt[n][j]);
        float se = 0.f;
        for (int j = 0; j < 16; ++j) se += __expf(Mt[n][j] - mx);
        la -= mx + logf(se);
        __syncthreads();
        Mt[n][m] = la; __syncthreads();
        mx = -1e30f;
        for (int j = 0; j < 16; ++j) mx = fmaxf(mx, Mt[j][m]);
        se = 0.f;
        for (int j = 0; j < 16; ++j) se += __expf(Mt[j][m] - mx);
        la -= mx + logf(se);
        __syncthreads();
    }
    Mt[n][m] = la; __syncthreads();
    if (m == 0) {
        int best = 0; float bv = Mt[n][0];
        for (int j = 1; j < 16; ++j) if (Mt[n][j] > bv) { bv = Mt[n][j]; best = j; }
        perm[b * NBLK + n] = best;
    }
}

// ---------------------------------------------------------------------------
// 4) Apply hard block permutation + f32 -> bf16
// ---------------------------------------------------------------------------
__global__ void permute_cvt_kernel(const float* __restrict__ x,
                                   const int* __restrict__ perm,
                                   bf16* __restrict__ xp) {
    size_t idx = ((size_t)blockIdx.x * blockDim.x + threadIdx.x) << 3;
    int r = (int)(idx >> 10);
    int d = (int)(idx & 1023);
    int sIdx = r >> 2, b = r & 3;
    int n = sIdx >> 8, i = sIdx & 255;
    int sp = perm[b * NBLK + n] * BS + i;
    const float* src = x + (((size_t)(sp * BATCH + b)) << 10) + d;
#pragma unroll
    for (int t = 0; t < 8; ++t) xp[idx + t] = (bf16)src[t];
}

// generic f32 -> bf16 converter (for weights)
__global__ void cvt_f32_bf16_kernel(const float* __restrict__ in, bf16* __restrict__ out, int n) {
    int i = blockIdx.x * blockDim.x + threadIdx.x;
    int stride = gridDim.x * blockDim.x;
    for (; i < n; i += stride) out[i] = (bf16)in[i];
}

// ---------------------------------------------------------------------------
// 5) bf16 WMMA GEMM with async-LDS double buffering:
//    Out[M,N] = A[M,K] * W[N,K]^T + bias
//    256 threads = 8 waves, WG tile 128x256, wave tile 64x64.
//    Per K-step (32): 6 async b128 copies per wave stage the next tile while
//    16 WMMAs consume the current one from LDS.
// ---------------------------------------------------------------------------
#define APAD 40   // 32 k-elements + pad (80B rows -> bank spread)
__global__ __launch_bounds__(256) void gemm_bf16_kernel(
    const bf16* __restrict__ A, const bf16* __restrict__ W,
    const float* __restrict__ bias,
    float* __restrict__ outF, bf16* __restrict__ outB,
    int M, int N, int K)
{
    __shared__ __align__(16) bf16 As[2][128][APAD];   // 20 KB
    __shared__ __align__(16) bf16 Bs[2][256][APAD];   // 40 KB

    int tid  = threadIdx.x;
    int lane = tid & 31;
    int wave = tid >> 5;                      // 0..7
    int wm = wave >> 2, wn = wave & 3;        // 2 x 4 wave grid
    int nBn  = N >> 8;                        // N / 256
    int bm = blockIdx.x / nBn, bn = blockIdx.x - bm * nBn;
    int rowBase = bm << 7;                    // 128-row tile
    int colBase = bn << 8;                    // 256-col tile
    int row0 = rowBase + (wm << 6);
    int col0 = colBase + (wn << 6);

    // --- per-thread async-copy slots: 16B chunk each -------------------------
    // A tile: 128 rows x 64B = 512 chunks -> 2 slots; B tile: 256 rows -> 4 slots
    int cA0 = tid, cA1 = 256 + tid;
    const bf16* gA0 = A + (size_t)(rowBase + (cA0 >> 2)) * K + ((cA0 & 3) << 3);
    const bf16* gA1 = A + (size_t)(rowBase + (cA1 >> 2)) * K + ((cA1 & 3) << 3);
    const bf16* gB[4];
    unsigned lB[2][4];
#pragma unroll
    for (int s = 0; s < 4; ++s) {
        int c = (s << 8) + tid;
        gB[s] = W + (size_t)(colBase + (c >> 2)) * K + ((c & 3) << 3);
        lB[0][s] = (unsigned)(size_t)&Bs[0][c >> 2][(c & 3) << 3];
        lB[1][s] = (unsigned)(size_t)&Bs[1][c >> 2][(c & 3) << 3];
    }
    unsigned lA0[2], lA1[2];
    lA0[0] = (unsigned)(size_t)&As[0][cA0 >> 2][(cA0 & 3) << 3];
    lA0[1] = (unsigned)(size_t)&As[1][cA0 >> 2][(cA0 & 3) << 3];
    lA1[0] = (unsigned)(size_t)&As[0][cA1 >> 2][(cA1 & 3) << 3];
    lA1[1] = (unsigned)(size_t)&As[1][cA1 >> 2][(cA1 & 3) << 3];

    auto issue_tile = [&](int buf, int k0) {
        async_cp16(lA0[buf], gA0 + k0);
        async_cp16(lA1[buf], gA1 + k0);
#pragma unroll
        for (int s = 0; s < 4; ++s) async_cp16(lB[buf][s], gB[s] + k0);
    };

    v8f acc[4][4] = {};
    const int nk = K >> 5;                    // K / 32
    issue_tile(0, 0);
    for (int kt = 0; kt < nk; ++kt) {
        int cur = kt & 1;
        if (kt + 1 < nk) {
            issue_tile(cur ^ 1, (kt + 1) << 5);
            asm volatile("s_wait_asynccnt 6" ::: "memory");   // current tile landed
        } else {
            asm volatile("s_wait_asynccnt 0" ::: "memory");
        }
        __syncthreads();

        v16bf a[4], w[4];
#pragma unroll
        for (int i = 0; i < 4; ++i)
            a[i] = frag_ld(&As[cur][(wm << 6) + 16 * i][0], APAD, lane);
#pragma unroll
        for (int j = 0; j < 4; ++j)
            w[j] = frag_ld(&Bs[cur][(wn << 6) + 16 * j][0], APAD, lane);
#pragma unroll
        for (int i = 0; i < 4; ++i)
#pragma unroll
            for (int j = 0; j < 4; ++j)
                acc[i][j] = __builtin_amdgcn_wmma_f32_16x16x32_bf16(
                    false, a[i], false, w[j], (short)0, acc[i][j], false, false);
        __syncthreads();
    }

    int lh = lane >> 4, lc = lane & 15;
#pragma unroll
    for (int j = 0; j < 4; ++j) {
        float bv = bias ? bias[col0 + 16 * j + lc] : 0.f;
#pragma unroll
        for (int i = 0; i < 4; ++i) {
#pragma unroll
            for (int g = 0; g < 8; ++g) {
                int r = row0 + 16 * i + g + (lh << 3);
                int c = col0 + 16 * j + lc;
                float v = acc[i][j][g] + bv;
                if (outF) outF[(size_t)r * N + c] = v;
                else      outB[(size_t)r * N + c] = (bf16)v;
            }
        }
    }
}

// ---------------------------------------------------------------------------
// 6) Block-local attention. One WG (4 waves) per (b, block n, head h, qchunk).
//    Each wave: 16 q rows; scores 16x256 live as 16 WMMA accumulators,
//    softmax in registers (half-wave shfl_xor reductions), P -> LDS,
//    V^T staged in LDS, P*V via WMMA, bf16 out (S*B, D) row-major.
// ---------------------------------------------------------------------------
#define KPAD 264
__global__ __launch_bounds__(128) void block_attn_kernel(
    const bf16* __restrict__ Q, const bf16* __restrict__ Kp,
    const bf16* __restrict__ V, bf16* __restrict__ AO)
{
    __shared__ __align__(16) bf16 vT[HDIM][KPAD];       // [hd col][key]
    __shared__ __align__(16) bf16 pbuf[4][16][KPAD];    // per wave: [q row][key]

    int tid = threadIdx.x;
    int lane = tid & 31, w = tid >> 5;
    int qc = blockIdx.x & 3;
    int h  = (blockIdx.x >> 2) & 15;
    int n  = (blockIdx.x >> 6) & 15;
    int b  = blockIdx.x >> 10;
    const int ldg = BATCH * DMODEL;                     // 4096: stride between seq rows

    // stage V^T (coalesced global reads, transposed LDS writes)
    for (int idx = tid; idx < BS * HDIM; idx += 128) {
        int k = idx >> 6, c = idx & 63;
        vT[c][k] = V[(size_t)((n * BS + k) * BATCH + b) * DMODEL + h * HDIM + c];
    }
    __syncthreads();

    int i0 = (qc << 6) + (w << 4);                      // first q row of this wave
    int lh = lane >> 4, lc = lane & 15;

    // Q fragments (16 rows x 64 K)
    v16bf aq[2];
#pragma unroll
    for (int kt = 0; kt < 2; ++kt)
        aq[kt] = frag_ld(Q + (size_t)((n * BS + i0) * BATCH + b) * DMODEL + h * HDIM + kt * 32,
                         ldg, lane);

    // scores: S[16 x 256] = Q * K^T
    v8f sacc[16];
#pragma unroll
    for (int nt = 0; nt < 16; ++nt) {
        v8f c = {};
#pragma unroll
        for (int kt = 0; kt < 2; ++kt) {
            v16bf bk = frag_ld(Kp + (size_t)((n * BS + nt * 16) * BATCH + b) * DMODEL
                                   + h * HDIM + kt * 32, ldg, lane);
            c = __builtin_amdgcn_wmma_f32_16x16x32_bf16(false, aq[kt], false, bk,
                                                        (short)0, c, false, false);
        }
        sacc[nt] = c;
    }

    // softmax per row; row m = g + 8*(lane>>4), col = nt*16 + (lane&15)
#pragma unroll
    for (int g = 0; g < 8; ++g) {
        float mx = -1e30f;
#pragma unroll
        for (int nt = 0; nt < 16; ++nt) {
            float sv = sacc[nt][g] * SCALEF;
            sacc[nt][g] = sv;
            mx = fmaxf(mx, sv);
        }
        mx = fmaxf(mx, __shfl_xor(mx, 1, 32));
        mx = fmaxf(mx, __shfl_xor(mx, 2, 32));
        mx = fmaxf(mx, __shfl_xor(mx, 4, 32));
        mx = fmaxf(mx, __shfl_xor(mx, 8, 32));
        float se = 0.f;
#pragma unroll
        for (int nt = 0; nt < 16; ++nt) se += __expf(sacc[nt][g] - mx);
        se += __shfl_xor(se, 1, 32);
        se += __shfl_xor(se, 2, 32);
        se += __shfl_xor(se, 4, 32);
        se += __shfl_xor(se, 8, 32);
        float inv = 1.f / se;
        int m = g + (lh << 3);
#pragma unroll
        for (int nt = 0; nt < 16; ++nt)
            pbuf[w][m][nt * 16 + lc] = (bf16)(__expf(sacc[nt][g] - mx) * inv);
    }
    __syncthreads();

    // Out[16 x 64] = P[16 x 256] * V[256 x 64]
    v8f oacc[4] = {};
#pragma unroll
    for (int kt = 0; kt < 8; ++kt) {
        v16bf ap = frag_ld(&pbuf[w][0][kt * 32], KPAD, lane);
#pragma unroll
        for (int ct = 0; ct < 4; ++ct) {
            v16bf bv = frag_ld(&vT[ct * 16][kt * 32], KPAD, lane);
            oacc[ct] = __builtin_amdgcn_wmma_f32_16x16x32_bf16(false, ap, false, bv,
                                                               (short)0, oacc[ct], false, false);
        }
    }

    // store bf16 attention output, row-major (S*B, D)
#pragma unroll
    for (int ct = 0; ct < 4; ++ct)
#pragma unroll
        for (int g = 0; g < 8; ++g) {
            int gr  = n * BS + i0 + g + (lh << 3);
            int col = h * HDIM + ct * 16 + lc;
            AO[(size_t)(gr * BATCH + b) * DMODEL + col] = (bf16)oacc[ct][g];
        }
}

// ---------------------------------------------------------------------------
// Workspace layout (needs ~74 MB). Q and K bf16 buffers live inside d_out
// (64 MB) and are fully consumed by attention before the final GEMM
// overwrites d_out with the f32 result.
// ---------------------------------------------------------------------------
static constexpr size_t OFF_PERM = 0;                       // 64 ints
static constexpr size_t OFF_XB   = 4096;                    // 64*1024 f32
static constexpr size_t OFF_QBLK = OFF_XB   + 262144;
static constexpr size_t OFF_KBLK = OFF_QBLK + 262144;
static constexpr size_t OFF_WQ   = 1048576;                 // bf16 1024x1024 each
static constexpr size_t OFF_WK   = OFF_WQ + 2097152;
static constexpr size_t OFF_WV   = OFF_WK + 2097152;
static constexpr size_t OFF_WO   = OFF_WV + 2097152;
static constexpr size_t OFF_XP   = OFF_WO + 2097152;        // bf16 16384x1024 (32 MB)
static constexpr size_t OFF_V    = OFF_XP + 33554432;       // bf16 16384x1024 (32 MB)
static constexpr size_t OFF_AO   = OFF_XP;                  // reuse x_perm region

extern "C" void kernel_launch(void* const* d_in, const int* in_sizes, int n_in,
                              void* d_out, int out_size, void* d_ws, size_t ws_size,
                              hipStream_t stream) {
    (void)in_sizes; (void)n_in; (void)out_size; (void)ws_size;
    const float* x  = (const float*)d_in[0];
    const float* Wq = (const float*)d_in[1];
    const float* bq = (const float*)d_in[2];
    const float* Wk = (const float*)d_in[3];
    const float* bk = (const float*)d_in[4];
    const float* Wv = (const float*)d_in[5];
    const float* bv = (const float*)d_in[6];
    const float* Wo = (const float*)d_in[7];
    const float* bo = (const float*)d_in[8];

    char* ws = (char*)d_ws;
    int*   perm = (int*)(ws + OFF_PERM);
    float* xb   = (float*)(ws + OFF_XB);
    float* qblk = (float*)(ws + OFF_QBLK);
    float* kblk = (float*)(ws + OFF_KBLK);
    bf16*  WQh  = (bf16*)(ws + OFF_WQ);
    bf16*  WKh  = (bf16*)(ws + OFF_WK);
    bf16*  WVh  = (bf16*)(ws + OFF_WV);
    bf16*  WOh  = (bf16*)(ws + OFF_WO);
    bf16*  XP   = (bf16*)(ws + OFF_XP);
    bf16*  Vb   = (bf16*)(ws + OFF_V);
    bf16*  AO   = (bf16*)(ws + OFF_AO);
    bf16*  Qb   = (bf16*)d_out;                          // 32 MB
    bf16*  Kb   = (bf16*)d_out + (size_t)NROWS * DMODEL; // next 32 MB

    const int WN = DMODEL * DMODEL;                      // weight elements

    // weights -> bf16
    cvt_f32_bf16_kernel<<<2048, 256, 0, stream>>>(Wq, WQh, WN);
    cvt_f32_bf16_kernel<<<2048, 256, 0, stream>>>(Wk, WKh, WN);
    cvt_f32_bf16_kernel<<<2048, 256, 0, stream>>>(Wv, WVh, WN);
    cvt_f32_bf16_kernel<<<2048, 256, 0, stream>>>(Wo, WOh, WN);

    // routing
    block_mean_kernel<<<BATCH * NBLK, 256, 0, stream>>>(x, xb);
    blk_proj_kernel<<<BATCH * NBLK, 256, 0, stream>>>(xb, Wq, bq, Wk, bk, qblk, kblk);
    sinkhorn_route_kernel<<<BATCH, 256, 0, stream>>>(qblk, kblk, perm);

    // permuted activations (bf16)
    permute_cvt_kernel<<<(NROWS * DMODEL) / (256 * 8), 256, 0, stream>>>(x, perm, XP);

    // QKV projections (WMMA, async-LDS double buffered)
    const int gemmGrid = (NROWS / 128) * (DMODEL / 256);   // 512
    gemm_bf16_kernel<<<gemmGrid, 256, 0, stream>>>(XP, WQh, bq, nullptr, Qb, NROWS, DMODEL, DMODEL);
    gemm_bf16_kernel<<<gemmGrid, 256, 0, stream>>>(XP, WKh, bk, nullptr, Kb, NROWS, DMODEL, DMODEL);
    gemm_bf16_kernel<<<gemmGrid, 256, 0, stream>>>(XP, WVh, bv, nullptr, Vb, NROWS, DMODEL, DMODEL);

    // block-local attention (WMMA)
    block_attn_kernel<<<BATCH * NBLK * NHEAD * 4, 128, 0, stream>>>(Qb, Kb, Vb, AO);

    // output projection -> f32 d_out (WMMA)
    gemm_bf16_kernel<<<gemmGrid, 256, 0, stream>>>(AO, WOh, bo, (float*)d_out, nullptr,
                                                   NROWS, DMODEL, DMODEL);
}
